// RGAT_45260365365584
// MI455X (gfx1250) — matrix operand order repaired and checked by reading, not compiled, not run on hardware
//
#include <hip/hip_runtime.h>
#include <hip/hip_bf16.h>

typedef __attribute__((ext_vector_type(16))) _Float16 v16h;
typedef __attribute__((ext_vector_type(8)))  float    v8f;

#define NN 100000
#define NE 600000
#define NG 64
#define FD 128

__device__ __forceinline__ float lrelu(float x) { return x > 0.f ? x : 0.2f * x; }

// order-preserving float -> uint key for atomicMax-based segment max
__device__ __forceinline__ unsigned fenc(float f) {
    unsigned u = __float_as_uint(f);
    return (u & 0x80000000u) ? ~u : (u | 0x80000000u);
}
__device__ __forceinline__ float fdec(unsigned k) {
    return (k & 0x80000000u) ? __uint_as_float(k & 0x7fffffffu) : __uint_as_float(~k);
}

__device__ __forceinline__ float wave_sum(float v) {
    #pragma unroll
    for (int o = 16; o; o >>= 1) v += __shfl_xor(v, o, 32);
    return v;
}

// ---------------------------------------------------------------------------
// Pack W_r [128x128 row-major f32] into WMMA B-fragment order (f16).
// pb index = ((r*32 + w*4 + c)*32 + lane)*16 + j ;  K = c*32 + (lane>=16)*8 + (j%8) + (j/8)*16,
// N = w*16 + lane%16.
// ---------------------------------------------------------------------------
__global__ void pack_b(const float* __restrict__ W0, const float* __restrict__ W1,
                       const float* __restrict__ W2, _Float16* __restrict__ pb) {
    int idx = blockIdx.x * blockDim.x + threadIdx.x;
    if (idx >= 3 * 16384) return;
    int r    = idx >> 14;
    int t    = idx & 16383;
    int j    = t & 15;
    int lane = (t >> 4) & 31;
    int wc   = t >> 9;           // w*4 + c, 0..31
    int c    = wc & 3;
    int wv   = wc >> 2;
    int k = c * 32 + ((lane >= 16) ? 8 : 0) + (j & 7) + ((j >> 3) << 4);
    int n = wv * 16 + (lane & 15);
    const float* W = (r == 0) ? W0 : (r == 1) ? W1 : W2;
    pb[idx] = (_Float16)W[k * FD + n];
}

// ---------------------------------------------------------------------------
// h = x @ W via v_wmma_f32_16x16x32_f16.  Each wave32 owns one 16-row slab:
// loads+converts its 4 A fragments (K=128) ONCE, then sweeps all 8 column
// tiles reusing them -> 32 WMMAs per wave, A traffic read exactly once.
// ---------------------------------------------------------------------------
__global__ void gemm_l1(const float* __restrict__ x, const _Float16* __restrict__ pb,
                        float* __restrict__ h) {
    int wv   = threadIdx.x >> 5;
    int lane = threadIdx.x & 31;
    int slab = blockIdx.x * 8 + wv;          // 16-row slab owned by this wave
    int m0   = slab * 16;
    if (m0 >= NN) return;

    const float* xr = x + (size_t)(m0 + (lane & 15)) * FD;
    int kbase = (lane < 16) ? 0 : 8;

    // A fragments for all of K: K(j) = c*32 + kbase + (j%8) + (j/8)*16
    v16h afrag[4];
    #pragma unroll
    for (int c = 0; c < 4; ++c) {
        const float* xk = xr + c * 32 + kbase;
        float4 f0 = *(const float4*)(xk + 0);
        float4 f1 = *(const float4*)(xk + 4);
        float4 f2 = *(const float4*)(xk + 16);
        float4 f3 = *(const float4*)(xk + 20);
        v16h a;
        a[0]=(_Float16)f0.x; a[1]=(_Float16)f0.y; a[2]=(_Float16)f0.z; a[3]=(_Float16)f0.w;
        a[4]=(_Float16)f1.x; a[5]=(_Float16)f1.y; a[6]=(_Float16)f1.z; a[7]=(_Float16)f1.w;
        a[8]=(_Float16)f2.x; a[9]=(_Float16)f2.y; a[10]=(_Float16)f2.z; a[11]=(_Float16)f2.w;
        a[12]=(_Float16)f3.x; a[13]=(_Float16)f3.y; a[14]=(_Float16)f3.z; a[15]=(_Float16)f3.w;
        afrag[c] = a;
    }

    int colb  = lane & 15;
    int rbase = m0 + ((lane >> 4) << 3);
    #pragma unroll
    for (int w = 0; w < 8; ++w) {
        v8f acc = {};
        #pragma unroll
        for (int c = 0; c < 4; ++c) {
            const _Float16* pbp = pb + ((size_t)((w * 4 + c) * 32 + lane) << 4);
            union { float4 q[2]; v16h v; } u;
            u.q[0] = ((const float4*)pbp)[0];
            u.q[1] = ((const float4*)pbp)[1];
            acc = __builtin_amdgcn_wmma_f32_16x16x32_f16(false, afrag[c], false, u.v,
                                                         (short)0, acc, false, false);
        }
        // D layout: VGPR v -> row m0 + v + (lane>=16)*8, col = 16*w + lane%16
        int col = w * 16 + colb;
        #pragma unroll
        for (int v = 0; v < 8; ++v)
            h[(size_t)(rbase + v) * FD + col] = acc[v];
    }
}

// a_s[i] = h[i,:].a_src ; a_d[i] = h[i,:].a_dst   (one wave32 per node)
__global__ void attn_coef(const float* __restrict__ h, const float* __restrict__ asrc,
                          const float* __restrict__ adst, float* __restrict__ a_s,
                          float* __restrict__ a_d, int n) {
    int t = blockIdx.x * blockDim.x + threadIdx.x;
    int i = t >> 5, lane = t & 31;
    if (i >= n) return;
    float4 hv = ((const float4*)(h + (size_t)i * FD))[lane];
    float4 sv = ((const float4*)asrc)[lane];
    float4 dv = ((const float4*)adst)[lane];
    float ps = hv.x * sv.x + hv.y * sv.y + hv.z * sv.z + hv.w * sv.w;
    float pd = hv.x * dv.x + hv.y * dv.y + hv.z * dv.z + hv.w * dv.w;
    ps = wave_sum(ps);
    pd = wave_sum(pd);
    if (lane == 0) { a_s[i] = ps; a_d[i] = pd; }
}

__global__ void m_init(const float* __restrict__ a_s, const float* __restrict__ a_d,
                       unsigned* __restrict__ mkey, int n) {
    int i = blockIdx.x * blockDim.x + threadIdx.x;
    if (i >= n) return;
    mkey[i] = fenc(lrelu(a_s[i] + a_d[i]));      // self-loop edge
}

__global__ void edge_max(const int* __restrict__ src, const int* __restrict__ dst,
                         const float* __restrict__ a_s, const float* __restrict__ a_d,
                         unsigned* __restrict__ mkey, int ne) {
    int e = blockIdx.x * blockDim.x + threadIdx.x;
    if (e >= ne) return;
    int s = src[e], d = dst[e];
    atomicMax(&mkey[d], fenc(lrelu(a_s[s] + a_d[d])));
}

__global__ void denom_init(const float* __restrict__ a_s, const float* __restrict__ a_d,
                           const unsigned* __restrict__ mkey, float* __restrict__ denom, int n) {
    int i = blockIdx.x * blockDim.x + threadIdx.x;
    if (i >= n) return;
    denom[i] = __expf(lrelu(a_s[i] + a_d[i]) - fdec(mkey[i]));
}

__global__ void edge_denom(const int* __restrict__ src, const int* __restrict__ dst,
                           const float* __restrict__ a_s, const float* __restrict__ a_d,
                           const unsigned* __restrict__ mkey, float* __restrict__ denom, int ne) {
    int e = blockIdx.x * blockDim.x + threadIdx.x;
    if (e >= ne) return;
    int s = src[e], d = dst[e];
    atomicAdd(&denom[d], __expf(lrelu(a_s[s] + a_d[d]) - fdec(mkey[d])));
}

// self-loop contribution (ordered before the atomic edge pass -> plain RMW)
__global__ void self_agg(const float* __restrict__ h, const float* __restrict__ a_s,
                         const float* __restrict__ a_d, const unsigned* __restrict__ mkey,
                         const float* __restrict__ denom, float* __restrict__ out, int n) {
    int t = blockIdx.x * blockDim.x + threadIdx.x;
    int i = t >> 5, q = t & 31;
    if (i >= n) return;
    float alpha = __expf(lrelu(a_s[i] + a_d[i]) - fdec(mkey[i])) / denom[i];
    float4 hv = ((const float4*)(h + (size_t)i * FD))[q];
    float4* o = (float4*)(out + (size_t)i * FD) + q;
    float4 ov = *o;
    ov.x += alpha * hv.x; ov.y += alpha * hv.y; ov.z += alpha * hv.z; ov.w += alpha * hv.w;
    *o = ov;
}

// wave32 per edge: gather h[src] as float4/lane, atomic scatter into out[dst] (L2-resident)
__global__ void edge_agg(const int* __restrict__ src, const int* __restrict__ dst,
                         const float* __restrict__ h, const float* __restrict__ a_s,
                         const float* __restrict__ a_d, const unsigned* __restrict__ mkey,
                         const float* __restrict__ denom, float* __restrict__ out, int ne) {
    int t = blockIdx.x * blockDim.x + threadIdx.x;
    int e = t >> 5, lane = t & 31;
    if (e >= ne) return;
    int s = src[e], d = dst[e];
    float alpha = __expf(lrelu(a_s[s] + a_d[d]) - fdec(mkey[d])) / denom[d];
    float4 hv = ((const float4*)(h + (size_t)s * FD))[lane];
    float* ob = out + (size_t)d * FD + lane * 4;
    atomicAdd(ob + 0, alpha * hv.x);
    atomicAdd(ob + 1, alpha * hv.y);
    atomicAdd(ob + 2, alpha * hv.z);
    atomicAdd(ob + 3, alpha * hv.w);
}

__global__ void bias_elu(float* __restrict__ out, const float* __restrict__ b0,
                         const float* __restrict__ b1, const float* __restrict__ b2,
                         int total) {
    int i = blockIdx.x * blockDim.x + threadIdx.x;
    if (i >= total) return;
    int c = i & (FD - 1);
    float v = out[i] + b0[c] + b1[c] + b2[c];
    out[i] = v > 0.f ? v : (__expf(v) - 1.f);
}

// layer 2 (dout = 1): h2[i] = x2[i,:] . W2[:,0]   (one wave32 per node)
__global__ void gemm_l2(const float* __restrict__ x2, const float* __restrict__ W2,
                        float* __restrict__ h2, int n) {
    int t = blockIdx.x * blockDim.x + threadIdx.x;
    int i = t >> 5, lane = t & 31;
    if (i >= n) return;
    float4 xv = ((const float4*)(x2 + (size_t)i * FD))[lane];
    float4 wv = ((const float4*)W2)[lane];
    float p = wave_sum(xv.x * wv.x + xv.y * wv.y + xv.z * wv.z + xv.w * wv.w);
    if (lane == 0) h2[i] = p;
}

__global__ void m2_init(const float* __restrict__ h2, const float* __restrict__ asc,
                        const float* __restrict__ adc, unsigned* __restrict__ mkey, int n) {
    int i = blockIdx.x * blockDim.x + threadIdx.x;
    if (i >= n) return;
    mkey[i] = fenc(lrelu(h2[i] * (asc[0] + adc[0])));
}

__global__ void edge2_max(const int* __restrict__ src, const int* __restrict__ dst,
                          const float* __restrict__ h2, const float* __restrict__ asc,
                          const float* __restrict__ adc, unsigned* __restrict__ mkey, int ne) {
    int e = blockIdx.x * blockDim.x + threadIdx.x;
    if (e >= ne) return;
    int s = src[e], d = dst[e];
    atomicMax(&mkey[d], fenc(lrelu(h2[s] * asc[0] + h2[d] * adc[0])));
}

__global__ void denom2_init(const float* __restrict__ h2, const float* __restrict__ asc,
                            const float* __restrict__ adc, const unsigned* __restrict__ mkey,
                            float* __restrict__ denom, int n) {
    int i = blockIdx.x * blockDim.x + threadIdx.x;
    if (i >= n) return;
    denom[i] = __expf(lrelu(h2[i] * (asc[0] + adc[0])) - fdec(mkey[i]));
}

__global__ void edge2_denom(const int* __restrict__ src, const int* __restrict__ dst,
                            const float* __restrict__ h2, const float* __restrict__ asc,
                            const float* __restrict__ adc, const unsigned* __restrict__ mkey,
                            float* __restrict__ denom, int ne) {
    int e = blockIdx.x * blockDim.x + threadIdx.x;
    if (e >= ne) return;
    int s = src[e], d = dst[e];
    atomicAdd(&denom[d], __expf(lrelu(h2[s] * asc[0] + h2[d] * adc[0]) - fdec(mkey[d])));
}

__global__ void self2_agg(const float* __restrict__ h2, const float* __restrict__ asc,
                          const float* __restrict__ adc, const unsigned* __restrict__ mkey,
                          const float* __restrict__ denom, float* __restrict__ out2, int n) {
    int i = blockIdx.x * blockDim.x + threadIdx.x;
    if (i >= n) return;
    float alpha = __expf(lrelu(h2[i] * (asc[0] + adc[0])) - fdec(mkey[i])) / denom[i];
    out2[i] += alpha * h2[i];
}

__global__ void edge2_agg(const int* __restrict__ src, const int* __restrict__ dst,
                          const float* __restrict__ h2, const float* __restrict__ asc,
                          const float* __restrict__ adc, const unsigned* __restrict__ mkey,
                          const float* __restrict__ denom, float* __restrict__ out2, int ne) {
    int e = blockIdx.x * blockDim.x + threadIdx.x;
    if (e >= ne) return;
    int s = src[e], d = dst[e];
    float alpha = __expf(lrelu(h2[s] * asc[0] + h2[d] * adc[0]) - fdec(mkey[d])) / denom[d];
    atomicAdd(&out2[d], alpha * h2[s]);
}

__global__ void pool(const float* __restrict__ out2, const int* __restrict__ batch,
                     const float* __restrict__ b0, const float* __restrict__ b1,
                     const float* __restrict__ b2, float* __restrict__ sums,
                     float* __restrict__ cnt, int n) {
    int i = blockIdx.x * blockDim.x + threadIdx.x;
    if (i >= n) return;
    int g = batch[i];
    atomicAdd(&sums[g], out2[i] + b0[0] + b1[0] + b2[0]);
    atomicAdd(&cnt[g], 1.f);
}

__global__ void finalize(const float* __restrict__ sums, const float* __restrict__ cnt,
                         float* __restrict__ outp) {
    int g = threadIdx.x;
    if (g < NG) outp[g] = sums[g] / fmaxf(cnt[g], 1.f);
}

// ---------------------------------------------------------------------------
extern "C" void kernel_launch(void* const* d_in, const int* in_sizes, int n_in,
                              void* d_out, int out_size, void* d_ws, size_t ws_size,
                              hipStream_t stream) {
    (void)in_sizes; (void)n_in; (void)out_size; (void)ws_size;
    const float* x     = (const float*)d_in[0];
    const int*   ei[3] = {(const int*)d_in[1], (const int*)d_in[2], (const int*)d_in[3]};
    const int*   batch = (const int*)d_in[4];
    // params: layer0 rel r at 5+4r: {W, a_src, a_dst, b}; layer1 rel r at 17+4r.

    char* wp = (char*)d_ws;
    auto take = [&](size_t bytes) { char* p = wp; wp += (bytes + 255) & ~(size_t)255; return p; };
    float*    h      = (float*)take((size_t)NN * FD * 4);
    float*    out1   = (float*)take((size_t)NN * FD * 4);   // layer-1 sum, becomes x2
    float*    a_s    = (float*)take((size_t)NN * 4);
    float*    a_d    = (float*)take((size_t)NN * 4);
    unsigned* mkey   = (unsigned*)take((size_t)NN * 4);
    float*    denom  = (float*)take((size_t)NN * 4);
    float*    h2     = (float*)take((size_t)NN * 4);
    unsigned* m2key  = (unsigned*)take((size_t)NN * 4);
    float*    denom2 = (float*)take((size_t)NN * 4);
    float*    out2   = (float*)take((size_t)NN * 4);
    float*    sums   = (float*)take(NG * 4);
    float*    cnt    = (float*)take(NG * 4);
    _Float16* pb     = (_Float16*)take((size_t)3 * 16384 * 2);

    hipMemsetAsync(out1, 0, (size_t)NN * FD * 4, stream);
    hipMemsetAsync(out2, 0, (size_t)NN * 4, stream);
    hipMemsetAsync(sums, 0, NG * 4, stream);
    hipMemsetAsync(cnt,  0, NG * 4, stream);

    pack_b<<<192, 256, 0, stream>>>((const float*)d_in[5], (const float*)d_in[9],
                                    (const float*)d_in[13], pb);

    const int nodeB  = (NN + 255) / 256;
    const int edgeB  = (NE + 255) / 256;
    const int nodeWB = NN * 32 / 256;          // wave-per-node
    const int edgeWB = NE * 32 / 256;          // wave-per-edge
    const int slabB  = (NN / 16 + 7) / 8;      // wave-per-16-row-slab, 8 waves/block

    // ---------------- layer 1 (HeteroConv sum over 3 relations) -------------
    for (int r = 0; r < 3; ++r) {
        const float* asr = (const float*)d_in[6 + r * 4];
        const float* adt = (const float*)d_in[7 + r * 4];
        const int*   src = ei[r];
        const int*   dst = ei[r] + NE;

        gemm_l1<<<slabB, 256, 0, stream>>>(x, pb + (size_t)r * 16384, h);
        attn_coef<<<nodeWB, 256, 0, stream>>>(h, asr, adt, a_s, a_d, NN);
        m_init<<<nodeB, 256, 0, stream>>>(a_s, a_d, mkey, NN);
        edge_max<<<edgeB, 256, 0, stream>>>(src, dst, a_s, a_d, mkey, NE);
        denom_init<<<nodeB, 256, 0, stream>>>(a_s, a_d, mkey, denom, NN);
        edge_denom<<<edgeB, 256, 0, stream>>>(src, dst, a_s, a_d, mkey, denom, NE);
        self_agg<<<nodeWB, 256, 0, stream>>>(h, a_s, a_d, mkey, denom, out1, NN);
        edge_agg<<<edgeWB, 256, 0, stream>>>(src, dst, h, a_s, a_d, mkey, denom, out1, NE);
    }
    bias_elu<<<(NN * FD + 255) / 256, 256, 0, stream>>>(
        out1, (const float*)d_in[8], (const float*)d_in[12], (const float*)d_in[16], NN * FD);

    // ---------------- layer 2 (dout = 1) ------------------------------------
    for (int r = 0; r < 3; ++r) {
        const float* W2  = (const float*)d_in[17 + r * 4];
        const float* asc = (const float*)d_in[18 + r * 4];
        const float* adc = (const float*)d_in[19 + r * 4];
        const int*   src = ei[r];
        const int*   dst = ei[r] + NE;

        gemm_l2<<<nodeWB, 256, 0, stream>>>(out1, W2, h2, NN);
        m2_init<<<nodeB, 256, 0, stream>>>(h2, asc, adc, m2key, NN);
        edge2_max<<<edgeB, 256, 0, stream>>>(src, dst, h2, asc, adc, m2key, NE);
        denom2_init<<<nodeB, 256, 0, stream>>>(h2, asc, adc, m2key, denom2, NN);
        edge2_denom<<<edgeB, 256, 0, stream>>>(src, dst, h2, asc, adc, m2key, denom2, NE);
        self2_agg<<<nodeB, 256, 0, stream>>>(h2, asc, adc, m2key, denom2, out2, NN);
        edge2_agg<<<edgeB, 256, 0, stream>>>(src, dst, h2, asc, adc, m2key, denom2, out2, NE);
    }

    // ---------------- global mean pool --------------------------------------
    pool<<<nodeB, 256, 0, stream>>>(out2, batch, (const float*)d_in[20],
                                    (const float*)d_in[24], (const float*)d_in[28],
                                    sums, cnt, NN);
    finalize<<<1, 64, 0, stream>>>(sums, cnt, (float*)d_out);
}
